// VectorQuantizer_28913719836762
// MI455X (gfx1250) — matrix operand order repaired and checked by reading, not compiled, not run on hardware
//
#include <hip/hip_runtime.h>
#include <hip/hip_bf16.h>
#include <math.h>

typedef __attribute__((ext_vector_type(16))) __bf16 v16bf;
typedef __attribute__((ext_vector_type(8)))  __bf16 v8bf;
typedef __attribute__((ext_vector_type(8)))  float  v8f;

#define NTOK   (64 * 4096)   // B*T
#define TILE   128           // tokens per tile
#define NTILES (NTOK / TILE) // 2048
#define NBLOCKS 512          // persistent-ish: 4 tiles per block
#define MDIM   32
#define HDIM   128
#define DDIM   64
#define KCB    512
#define LNEPS  1e-5f
#define BETA   0.25f

// ---- dynamic LDS layout (byte offsets) ----
// sZf   : float[128*64]        @ 0        (32768)  z tile (pre-LN then post-LN fp32)
// sC2   : float[512]           @ 32768    (2048)   per-code squared norms
// sCode : int[128]             @ 34816    (512)    argmin codes per token
// sW1f  : bf16[8*32*16]        @ 35328    (8192)   W1 B-fragments [nt][lane][e]
// sW2f  : bf16[16*32*16]       @ 43520    (16384)  W2 B-fragments [ks*4+nt][lane][e]
// sCBf  : bf16[64*32*16]       @ 59904    (65536)  codebook B-fragments [nt*2+ks][lane][e]
// sH    : bf16[128*128]        @ 125440   (32768)  GELU(hidden) tile, row-major
#define SMEM_BYTES 158208

__device__ __forceinline__ v8f wmma_bf16(v16bf a, v16bf b, v8f c) {
    return __builtin_amdgcn_wmma_f32_16x16x32_bf16(
        /*neg_a=*/false, a, /*neg_b=*/false, b,
        /*c_mod=*/(short)0, c, /*reuse_a=*/false, /*reuse_b=*/false);
}

__device__ __forceinline__ v16bf cat8(v8bf lo, v8bf hi) {
    return __builtin_shufflevector(lo, hi, 0,1,2,3,4,5,6,7,8,9,10,11,12,13,14,15);
}

__global__ void vq_zero_ws(unsigned int* hist, float* sumsq) {
    int t = threadIdx.x;
    if (t < KCB) hist[t] = 0u;
    if (t == 0) *sumsq = 0.0f;
}

__global__ __launch_bounds__(256)
void vq_fused_kernel(const float* __restrict__ x,
                     const float* __restrict__ W1, const float* __restrict__ b1,
                     const float* __restrict__ W2, const float* __restrict__ b2,
                     const float* __restrict__ ln_g, const float* __restrict__ ln_b,
                     const float* __restrict__ CB,
                     float* __restrict__ out,
                     unsigned int* __restrict__ hist,
                     float* __restrict__ sumsq) {
    extern __shared__ char smem[];
    float*  sZf   = (float*)(smem + 0);
    float*  sC2   = (float*)(smem + 32768);
    int*    sCode = (int*)  (smem + 34816);
    __bf16* sW1f  = (__bf16*)(smem + 35328);
    __bf16* sW2f  = (__bf16*)(smem + 43520);
    __bf16* sCBf  = (__bf16*)(smem + 59904);
    __bf16* sH    = (__bf16*)(smem + 125440);

    const int tid  = threadIdx.x;
    const int wave = tid >> 5;
    const int lane = tid & 31;
    const int half = lane >> 4;
    const int idx  = lane & 15;
    const int rowBase = wave * 16;                 // M-tile within the token tile

    float* zE     = out;                                        // (N,64)
    float* zQ     = out + (size_t)NTOK * DDIM;                  // (N,64)
    float* codesO = out + 2 * (size_t)NTOK * DDIM;              // (N,) as float

    // ---- Stage 0 (once per block): swizzle weights + codebook into LDS ----
    // B-fragment element mapping: lane=(h2,ix), element e -> k = (e>>3)*16 + h2*8 + (e&7), col n = ix
    for (int i = tid; i < 8 * 32 * 16; i += 256) {      // W1: (32 x 128), 8 N-tiles
        int e = i & 15, ln = (i >> 4) & 31, nt = i >> 9;
        int h2 = ln >> 4, ix = ln & 15;
        sW1f[i] = (__bf16)W1[(((e >> 3) << 4) + (h2 << 3) + (e & 7)) * HDIM + nt * 16 + ix];
    }
    for (int i = tid; i < 16 * 32 * 16; i += 256) {     // W2: (128 x 64), 4 ks x 4 nt
        int e = i & 15, ln = (i >> 4) & 31, t = i >> 9;
        int ks = t >> 2, nt = t & 3;
        int h2 = ln >> 4, ix = ln & 15;
        sW2f[i] = (__bf16)W2[(ks * 32 + ((e >> 3) << 4) + (h2 << 3) + (e & 7)) * DDIM + nt * 16 + ix];
    }
    for (int i = tid; i < 64 * 32 * 16; i += 256) {     // codebook as B: B[k][n]=CB[n][k]
        int e = i & 15, ln = (i >> 4) & 31, t = i >> 9;
        int nt = t >> 1, ks = t & 1;
        int h2 = ln >> 4, ix = ln & 15;
        sCBf[i] = (__bf16)CB[(nt * 16 + ix) * DDIM + ks * 32 + ((e >> 3) << 4) + (h2 << 3) + (e & 7)];
    }
    for (int c = tid; c < KCB; c += 256) {
        float s = 0.0f;
        #pragma unroll 8
        for (int d = 0; d < DDIM; ++d) { float v = CB[c * DDIM + d]; s += v * v; }
        sC2[c] = s;
    }
    __syncthreads();

    // ---- Persistent tile loop: amortize codebook staging over 4 tiles ----
    for (int tile = blockIdx.x; tile < NTILES; tile += gridDim.x) {
        const size_t tokBase = (size_t)tile * TILE;

        // ---- GEMM1: (16x32) @ (32x128), K=32 = one WMMA step ----
        // A-fragment: e<8 -> k = half*8+e ; e>=8 -> k = 16+half*8+(e-8)
        v16bf a;
        {
            const float* xrow = x + (tokBase + rowBase + idx) * MDIM + (half << 3);
            #pragma unroll
            for (int e = 0; e < 8; ++e) {
                a[e]     = (__bf16)xrow[e];
                a[e + 8] = (__bf16)xrow[16 + e];
            }
        }
        v8f accH[8];
        #pragma unroll
        for (int nt = 0; nt < 8; ++nt) {
            v16bf b = *(const v16bf*)(sW1f + ((nt << 5) + lane) * 16);
            v8f c = {};
            accH[nt] = wmma_bf16(a, b, c);
        }
        // bias + exact GELU, store h tile (bf16, row-major)
        #pragma unroll
        for (int nt = 0; nt < 8; ++nt) {
            int n = nt * 16 + idx;
            float bias = b1[n];
            #pragma unroll
            for (int j = 0; j < 8; ++j) {
                int m = rowBase + j + (half << 3);
                float v = accH[nt][j] + bias;
                float g = 0.5f * v * (1.0f + erff(v * 0.70710678118654752f));
                sH[m * HDIM + n] = (__bf16)g;
            }
        }
        // also orders prev-iteration epilogue (reads sZf) before GEMM2 writes sZf
        __syncthreads();

        // ---- GEMM2: (16x128) @ (128x64), K=128 = 4 WMMA steps ----
        v8f accZ[4];
        #pragma unroll
        for (int nt = 0; nt < 4; ++nt) accZ[nt] = (v8f){};
        #pragma unroll
        for (int ks = 0; ks < 4; ++ks) {
            const __bf16* hrow = sH + (rowBase + idx) * HDIM + ks * 32 + (half << 3);
            v16bf a2 = cat8(*(const v8bf*)hrow, *(const v8bf*)(hrow + 16));
            #pragma unroll
            for (int nt = 0; nt < 4; ++nt) {
                v16bf b = *(const v16bf*)(sW2f + (((ks << 2) + nt) << 5) * 16 + lane * 16);
                accZ[nt] = wmma_bf16(a2, b, accZ[nt]);
            }
        }
        #pragma unroll
        for (int nt = 0; nt < 4; ++nt) {
            int n = nt * 16 + idx;
            float bias = b2[n];
            #pragma unroll
            for (int j = 0; j < 8; ++j) {
                int m = rowBase + j + (half << 3);
                sZf[m * DDIM + n] = accZ[nt][j] + bias;
            }
        }
        __syncthreads();

        // ---- LayerNorm per token (fp32), write z_e to HBM, keep normalized in LDS ----
        if (tid < TILE) {
            float* zr = sZf + tid * DDIM;
            float mu = 0.0f;
            #pragma unroll 8
            for (int c = 0; c < DDIM; ++c) mu += zr[c];
            mu *= (1.0f / DDIM);
            float var = 0.0f;
            #pragma unroll 8
            for (int c = 0; c < DDIM; ++c) { float d = zr[c] - mu; var += d * d; }
            var *= (1.0f / DDIM);
            float rs = rsqrtf(var + LNEPS);
            float* ze = zE + (tokBase + tid) * DDIM;
            #pragma unroll 8
            for (int c = 0; c < DDIM; ++c) {
                float v = (zr[c] - mu) * rs * ln_g[c] + ln_b[c];
                zr[c] = v;
                ze[c] = v;
            }
        }
        __syncthreads();

        // ---- GEMM3: (16x64) @ (64x512) vs codebook, K=64 = 2 WMMA steps/tile ----
        // argmin(dist) == argmin(||c||^2 - 2 z.c)  (||z||^2 is row-constant)
        float bestD[8];
        int   bestI[8];
        #pragma unroll
        for (int j = 0; j < 8; ++j) { bestD[j] = 3.4e38f; bestI[j] = 0; }

        v16bf aZ[2];
        #pragma unroll
        for (int ks = 0; ks < 2; ++ks) {
            const float* zrow = sZf + (rowBase + idx) * DDIM + ks * 32 + (half << 3);
            #pragma unroll
            for (int e = 0; e < 8; ++e) {
                aZ[ks][e]     = (__bf16)zrow[e];
                aZ[ks][e + 8] = (__bf16)zrow[16 + e];
            }
        }
        for (int nt = 0; nt < KCB / 16; ++nt) {
            v8f acc = {};
            #pragma unroll
            for (int ks = 0; ks < 2; ++ks) {
                v16bf b = *(const v16bf*)(sCBf + (((nt << 1) + ks) << 5) * 16 + lane * 16);
                acc = wmma_bf16(aZ[ks], b, acc);
            }
            int n = nt * 16 + idx;
            float c2 = sC2[n];
            #pragma unroll
            for (int j = 0; j < 8; ++j) {
                float d = c2 - 2.0f * acc[j];
                if (d < bestD[j]) { bestD[j] = d; bestI[j] = n; }
            }
        }
        // cross-lane argmin over the 16 column-lanes of each half-wave
        #pragma unroll
        for (int j = 0; j < 8; ++j) {
            float m = bestD[j]; int mi = bestI[j];
            #pragma unroll
            for (int off = 1; off < 16; off <<= 1) {
                float om = __shfl_xor(m, off, 32);
                int   oi = __shfl_xor(mi, off, 32);
                if (om < m || (om == m && oi < mi)) { m = om; mi = oi; }
            }
            if (idx == 0) sCode[rowBase + j + (half << 3)] = mi;
        }
        __syncthreads();

        // ---- Epilogue: codes, z_q gather, loss + histogram accumulation ----
        if (tid < TILE) {
            int code = sCode[tid];
            codesO[tokBase + tid] = (float)code;
            const float* cb = CB + (size_t)code * DDIM;
            float* zq = zQ + (tokBase + tid) * DDIM;
            const float* zr = sZf + tid * DDIM;
            float ss = 0.0f;
            #pragma unroll 8
            for (int c = 0; c < DDIM; ++c) {
                float q = cb[c];
                zq[c] = q;                   // z_q_st == z_q numerically (straight-through)
                float d = q - zr[c];
                ss += d * d;
            }
            atomicAdd(sumsq, ss);
            atomicAdd(&hist[code], 1u);
        }
        // NOTE: no barrier needed here; next iteration's first touch of shared state
        // is sH (GEMM1), which the epilogue never reads, and the post-GEMM1 barrier
        // orders this epilogue before the next sZf overwrite.
    }
}

__global__ void vq_finalize(const unsigned int* __restrict__ hist,
                            const float* __restrict__ sumsq,
                            float* __restrict__ out) {
    __shared__ float red[KCB];
    int t = threadIdx.x;
    float p = (float)hist[t] / (float)NTOK;
    red[t] = -p * logf(p + 1e-10f);
    __syncthreads();
    for (int s = KCB / 2; s > 0; s >>= 1) {
        if (t < s) red[t] += red[t + s];
        __syncthreads();
    }
    if (t == 0) {
        float* scal = out + 2 * (size_t)NTOK * DDIM + NTOK;
        float loss = *sumsq / ((float)NTOK * (float)DDIM);
        scal[0] = expf(red[0]);      // perplexity
        scal[1] = loss;              // codebook loss
        scal[2] = BETA * loss;       // commitment loss (same value * beta)
    }
}

extern "C" void kernel_launch(void* const* d_in, const int* in_sizes, int n_in,
                              void* d_out, int out_size, void* d_ws, size_t ws_size,
                              hipStream_t stream) {
    const float* x    = (const float*)d_in[0];
    const float* W1   = (const float*)d_in[1];
    const float* b1   = (const float*)d_in[2];
    const float* W2   = (const float*)d_in[3];
    const float* b2   = (const float*)d_in[4];
    const float* ln_g = (const float*)d_in[5];
    const float* ln_b = (const float*)d_in[6];
    const float* CB   = (const float*)d_in[7];
    float* out = (float*)d_out;

    unsigned int* hist = (unsigned int*)d_ws;
    float* sumsq = (float*)(hist + KCB);

    vq_zero_ws<<<1, KCB, 0, stream>>>(hist, sumsq);

    dim3 grid(NBLOCKS);       // each block loops over NTILES/NBLOCKS = 4 tiles
    dim3 block(256);          // 8 wave32s
    vq_fused_kernel<<<grid, block, SMEM_BYTES, stream>>>(
        x, W1, b1, W2, b2, ln_g, ln_b, CB, out, hist, sumsq);

    vq_finalize<<<1, KCB, 0, stream>>>(hist, sumsq, out);
}
